// DynAGCN_13709535609425
// MI455X (gfx1250) — matrix operand order repaired and checked by reading, not compiled, not run on hardware
//
#include <hip/hip_runtime.h>

#define DEV __device__ __forceinline__

typedef __attribute__((ext_vector_type(16))) __bf16 v16bf;
typedef __attribute__((ext_vector_type(8)))  __bf16 v8bf;
typedef __attribute__((ext_vector_type(8)))  float  v8f;

namespace {
constexpr int Bn  = 16;       // batch
constexpr int Np  = 1024;     // nodes (H*W)
constexpr int Ed  = 64;       // embedding dim (= C_IN = C_OUT)
constexpr int Ch  = 32;       // fc hidden
constexpr int ROWS = Bn * Np; // 16384
}

DEV int lane_id() { return threadIdx.x & 31; }
DEV int wave_id() { return threadIdx.x >> 5; }

DEV float frcp(float x) { return __builtin_amdgcn_rcpf(x); }          // v_rcp_f32
DEV float sigm(float x) { return frcp(1.f + __expf(-x)); }            // exp + rcp
DEV float tanh_fast(float x) {                                        // 1 - 2/(e^2x+1)
  return 1.f - 2.f * frcp(__expf(2.f * x) + 1.f);
}

DEV void wait_ds() { asm volatile("s_wait_dscnt 0" ::: "memory"); }

DEV v8f wmma_bf(v16bf a, v16bf b, v8f c) {
  return __builtin_amdgcn_wmma_f32_16x16x32_bf16(false, a, false, b, (short)0, c, false, false);
}

// ---- fragment gathers for WMMA (wave32); A and B share contiguous k = 16*(lane/16) + e ----
// All bf16 sources are laid out so each lane's elements are contiguous:
// two 16B vector loads per fragment.

// A fragment [16 rows x 32 K], row-major bf16, row stride ld (elements, mult of 8)
DEV v16bf ldA16(const __bf16* p, int ld) {
  int l = lane_id(); int r = l & 15; int kb = (l < 16) ? 0 : 16;
  const __bf16* q = p + r * ld + kb;
  v8bf lo = *(const v8bf*)q;
  v8bf hi = *(const v8bf*)(q + 8);
  v16bf a;
#pragma unroll
  for (int e = 0; e < 8; ++e) { a[e] = lo[e]; a[e + 8] = hi[e]; }
  return a;
}
// B fragment [32 K x 16 N] from BT-layout source (row-major [N][K], stride ld)
DEV v16bf ldBT16(const __bf16* p, int ld) {
  int l = lane_id(); int c = l & 15; int kb = (l < 16) ? 0 : 16;
  const __bf16* q = p + c * ld + kb;
  v8bf lo = *(const v8bf*)q;
  v8bf hi = *(const v8bf*)(q + 8);
  v16bf b;
#pragma unroll
  for (int e = 0; e < 8; ++e) { b[e] = lo[e]; b[e + 8] = hi[e]; }
  return b;
}
// B fragment from LDS row-major [K][N] (strided; used for the SpMM panel)
DEV v16bf ldB_lds(const __bf16* p, int ld) {
  int l = lane_id(); int c = l & 15; int kb = (l < 16) ? 0 : 16;
  v16bf b;
#pragma unroll
  for (int e = 0; e < 16; ++e) b[e] = p[(kb + e) * ld + c];
  return b;
}

// ====================================================================
// Setup: f32 -> bf16 conversion (plain and transposed)
// ====================================================================
__global__ __launch_bounds__(256) void k_cvt(const float* __restrict__ s,
                                             __bf16* __restrict__ d, int n) {
  int i = blockIdx.x * 256 + threadIdx.x;
  if (i < n) d[i] = (__bf16)s[i];
}
// src f32 [R][C] -> dst bf16 [C][R]
__global__ __launch_bounds__(256) void k_cvt_t(const float* __restrict__ s,
                                               __bf16* __restrict__ d, int R, int C) {
  int i = blockIdx.x * 256 + threadIdx.x;
  if (i < R * C) {
    int r = i / C, c = i - r * C;
    d[(size_t)c * R + r] = (__bf16)s[i];
  }
}

// ====================================================================
// Kernel 1: fc stack + memory attention -> dyn (bf16)
// 256 blocks x 128 threads; each wave owns 16 rows end-to-end.
// ====================================================================
__global__ __launch_bounds__(128) void k_embed(
    const __bf16* __restrict__ xr,                                   // [ROWS][64]
    const float* __restrict__ b1, const float* __restrict__ b2, const float* __restrict__ b3,
    const __bf16* __restrict__ w1T,  // [32][64]
    const __bf16* __restrict__ w2T,  // [32][32]
    const __bf16* __restrict__ w3T,  // [64][32]
    const __bf16* __restrict__ WqT,  // [64][64]
    const __bf16* __restrict__ memb, // Mem bf16 [128][64] (BT layout of Mem^T)
    const __bf16* __restrict__ memT, // Mem^T bf16 [64][128] (BT layout of Mem)
    __bf16* __restrict__ dyn) {
  __shared__ __bf16 act[4][16][136];
  const int LD = 136;
  int w = wave_id(), l = lane_id();
  int row0 = blockIdx.x * 64 + w * 16;
  int col = l & 15, mb = (l < 16) ? 0 : 8;
  __bf16* A = &act[w][0][0];

  // h1 = sigmoid(X @ w1 + b1)  -> act cols [0,32)
  v16bf xa0 = ldA16(xr + (size_t)row0 * Ed, Ed);
  v16bf xa1 = ldA16(xr + (size_t)row0 * Ed + 32, Ed);
#pragma unroll
  for (int nt = 0; nt < 2; ++nt) {
    v8f acc = {};
    acc = wmma_bf(xa0, ldBT16(w1T + nt * 16 * Ed, Ed), acc);
    acc = wmma_bf(xa1, ldBT16(w1T + nt * 16 * Ed + 32, Ed), acc);
    float bia = b1[nt * 16 + col];
#pragma unroll
    for (int v = 0; v < 8; ++v) A[(mb + v) * LD + nt * 16 + col] = (__bf16)sigm(acc[v] + bia);
  }
  wait_ds();
  // h2 = sigmoid(h1 @ w2 + b2) -> act cols [32,64)
  {
    v16bf ha = ldA16(A, LD);
#pragma unroll
    for (int nt = 0; nt < 2; ++nt) {
      v8f acc = {};
      acc = wmma_bf(ha, ldBT16(w2T + nt * 16 * Ch, Ch), acc);
      float bia = b2[nt * 16 + col];
#pragma unroll
      for (int v = 0; v < 8; ++v) A[(mb + v) * LD + 32 + nt * 16 + col] = (__bf16)sigm(acc[v] + bia);
    }
  }
  wait_ds();
  // x_e = h2 @ w3 + b3 -> act cols [64,128)
  {
    v16bf ha = ldA16(A + 32, LD);
#pragma unroll
    for (int nt = 0; nt < 4; ++nt) {
      v8f acc = {};
      acc = wmma_bf(ha, ldBT16(w3T + nt * 16 * Ch, Ch), acc);
      float bia = b3[nt * 16 + col];
#pragma unroll
      for (int v = 0; v < 8; ++v) A[(mb + v) * LD + 64 + nt * 16 + col] = (__bf16)(acc[v] + bia);
    }
  }
  wait_ds();
  // query = x_e @ Wq -> act cols [0,64)
  {
    v16bf e0 = ldA16(A + 64, LD), e1 = ldA16(A + 96, LD);
#pragma unroll
    for (int nt = 0; nt < 4; ++nt) {
      v8f acc = {};
      acc = wmma_bf(e0, ldBT16(WqT + nt * 16 * Ed, Ed), acc);
      acc = wmma_bf(e1, ldBT16(WqT + nt * 16 * Ed + 32, Ed), acc);
#pragma unroll
      for (int v = 0; v < 8; ++v) A[(mb + v) * LD + nt * 16 + col] = (__bf16)acc[v];
    }
  }
  wait_ds();
  // scores = query @ Mem^T (in registers), row softmax over 128
  v8f sc[8];
  {
    v16bf q0 = ldA16(A, LD), q1 = ldA16(A + 32, LD);
#pragma unroll
    for (int mt = 0; mt < 8; ++mt) {
      v8f acc = {};
      acc = wmma_bf(q0, ldBT16(memb + (size_t)(mt * 16) * Ed, Ed), acc);
      acc = wmma_bf(q1, ldBT16(memb + (size_t)(mt * 16) * Ed + 32, Ed), acc);
      sc[mt] = acc;
    }
  }
#pragma unroll
  for (int v = 0; v < 8; ++v) {
    float mx = -1e30f;
#pragma unroll
    for (int mt = 0; mt < 8; ++mt) mx = fmaxf(mx, sc[mt][v]);
#pragma unroll
    for (int m2 = 1; m2 < 16; m2 <<= 1) mx = fmaxf(mx, __shfl_xor(mx, m2, 32));
    float s = 0.f;
#pragma unroll
    for (int mt = 0; mt < 8; ++mt) { float e = __expf(sc[mt][v] - mx); sc[mt][v] = e; s += e; }
#pragma unroll
    for (int m2 = 1; m2 < 16; m2 <<= 1) s += __shfl_xor(s, m2, 32);
    float inv = frcp(s);
#pragma unroll
    for (int mt = 0; mt < 8; ++mt) A[(mb + v) * LD + mt * 16 + col] = (__bf16)(sc[mt][v] * inv);
  }
  wait_ds();
  // dyn = att @ Mem  (BT layout of Mem is memT [64][128])
  {
    v16bf at[4];
#pragma unroll
    for (int ks = 0; ks < 4; ++ks) at[ks] = ldA16(A + ks * 32, LD);
#pragma unroll
    for (int nt = 0; nt < 4; ++nt) {
      v8f acc = {};
#pragma unroll
      for (int ks = 0; ks < 4; ++ks)
        acc = wmma_bf(at[ks], ldBT16(memT + (size_t)(nt * 16) * 128 + ks * 32, 128), acc);
#pragma unroll
      for (int v = 0; v < 8; ++v)
        dyn[(size_t)(row0 + mb + v) * Ed + nt * 16 + col] = (__bf16)acc[v];
    }
  }
}

// ====================================================================
// Kernel 2: per-node LayerNorm + node-side GEMMs
// ====================================================================
__global__ __launch_bounds__(128) void k_node(
    const float* __restrict__ emb, const __bf16* __restrict__ emb_bf,
    const float* __restrict__ ln_g, const float* __restrict__ ln_b,
    const __bf16* __restrict__ w_wT, const float* __restrict__ w_b,
    const __bf16* __restrict__ ur_wT, const float* __restrict__ ur_b,
    const __bf16* __restrict__ wu_wT, const float* __restrict__ wu_b,
    const __bf16* __restrict__ bpT,
    float* __restrict__ ne_o, float* __restrict__ ne_res_o,
    float* __restrict__ u_r_o, float* __restrict__ w_u_o, float* __restrict__ bias_n_o) {
  __shared__ __bf16 act[4][16][72];
  const int LD = 72;
  int w = wave_id(), l = lane_id();
  int n0 = blockIdx.x * 64 + w * 16;
  int r = l & 15, half = l >> 4;
  // LayerNorm: 2 lanes per row, 32 elements each, combine via shfl_xor(16)
  const float* src = emb + (size_t)(n0 + r) * Ed + half * 32;
  float vals[32]; float s = 0.f;
#pragma unroll
  for (int j = 0; j < 32; ++j) { vals[j] = src[j]; s += vals[j]; }
  s += __shfl_xor(s, 16, 32);
  float mu = s * (1.f / 64.f);
  float q = 0.f;
#pragma unroll
  for (int j = 0; j < 32; ++j) { float d = vals[j] - mu; q += d * d; }
  q += __shfl_xor(q, 16, 32);
  float inv = rsqrtf(q * (1.f / 64.f) + 1e-5f);
#pragma unroll
  for (int j = 0; j < 32; ++j) {
    int c = half * 32 + j;
    float nv = (vals[j] - mu) * inv * ln_g[c] + ln_b[c];
    act[w][r][c] = (__bf16)nv;
    ne_o[(size_t)(n0 + r) * Ed + c] = nv;
  }
  wait_ds();
  int col = l & 15, mb = (l < 16) ? 0 : 8;
  v16bf a0 = ldA16(&act[w][0][0], LD), a1 = ldA16(&act[w][0][32], LD);
#pragma unroll
  for (int nt = 0; nt < 4; ++nt) {
    int c = nt * 16 + col;
    v8f ar = {}, au = {}, aw = {};
    ar = wmma_bf(a0, ldBT16(w_wT + nt * 16 * Ed, Ed), ar);
    ar = wmma_bf(a1, ldBT16(w_wT + nt * 16 * Ed + 32, Ed), ar);
    au = wmma_bf(a0, ldBT16(ur_wT + nt * 16 * Ed, Ed), au);
    au = wmma_bf(a1, ldBT16(ur_wT + nt * 16 * Ed + 32, Ed), au);
    aw = wmma_bf(a0, ldBT16(wu_wT + nt * 16 * Ed, Ed), aw);
    aw = wmma_bf(a1, ldBT16(wu_wT + nt * 16 * Ed + 32, Ed), aw);
#pragma unroll
    for (int v = 0; v < 8; ++v) {
      size_t gi = (size_t)(n0 + mb + v) * Ed + c;
      ne_res_o[gi] = ar[v] + w_b[c] + (float)act[w][mb + v][c];
      u_r_o[gi]    = au[v] + ur_b[c];
      w_u_o[gi]    = aw[v] + wu_b[c];
    }
  }
  // bias_n = node_emb @ bias_pool (raw embedding)
  v16bf g0 = ldA16(emb_bf + (size_t)n0 * Ed, Ed), g1 = ldA16(emb_bf + (size_t)n0 * Ed + 32, Ed);
#pragma unroll
  for (int nt = 0; nt < 4; ++nt) {
    v8f acc = {};
    acc = wmma_bf(g0, ldBT16(bpT + nt * 16 * Ed, Ed), acc);
    acc = wmma_bf(g1, ldBT16(bpT + nt * 16 * Ed + 32, Ed), acc);
#pragma unroll
    for (int v = 0; v < 8; ++v)
      bias_n_o[(size_t)(n0 + mb + v) * Ed + nt * 16 + col] = acc[v];
  }
}

// ====================================================================
// Kernel 3: gated fusion -> comb (bf16)
// ====================================================================
__global__ __launch_bounds__(128) void k_comb(
    const __bf16* __restrict__ dyn, const float* __restrict__ ne,
    const float* __restrict__ ne_res, const float* __restrict__ u_r,
    const float* __restrict__ w_u,
    const __bf16* __restrict__ transT,
    const __bf16* __restrict__ wr_wT, const float* __restrict__ wr_b,
    const __bf16* __restrict__ wh_wT, const float* __restrict__ wh_b,
    __bf16* __restrict__ comb) {
  int w = wave_id(), l = lane_id();
  int row0 = blockIdx.x * 64 + w * 16;
  int n0 = row0 & (Np - 1);
  int col = l & 15, mb = (l < 16) ? 0 : 8;
  const __bf16* D = dyn + (size_t)row0 * Ed;
  v16bf d0 = ldA16(D, Ed), d1 = ldA16(D + 32, Ed);
#pragma unroll
  for (int nt = 0; nt < 4; ++nt) {
    v8f at = {}, ar = {}, ah = {};
    at = wmma_bf(d0, ldBT16(transT + nt * 16 * Ed, Ed), at);
    at = wmma_bf(d1, ldBT16(transT + nt * 16 * Ed + 32, Ed), at);
    ar = wmma_bf(d0, ldBT16(wr_wT + nt * 16 * Ed, Ed), ar);
    ar = wmma_bf(d1, ldBT16(wr_wT + nt * 16 * Ed + 32, Ed), ar);
    ah = wmma_bf(d0, ldBT16(wh_wT + nt * 16 * Ed, Ed), ah);
    ah = wmma_bf(d1, ldBT16(wh_wT + nt * 16 * Ed + 32, Ed), ah);
#pragma unroll
    for (int v = 0; v < 8; ++v) {
      int rr = mb + v, c = nt * 16 + col;
      size_t ni = (size_t)(n0 + rr) * Ed + c;
      float dv = (float)D[rr * Ed + c];
      float z  = sigm(ne_res[ni] + dv + at[v]);
      float rg = sigm(ar[v] + wr_b[c] + u_r[ni]);
      float hh = tanh_fast(ah[v] + wh_b[c] + rg * w_u[ni]);
      comb[(size_t)(row0 + rr) * Ed + c] = (__bf16)(z * ne[ni] + (1.f - z) * hh);
    }
  }
}

// ====================================================================
// Kernel 4: support = softmax(relu(comb @ comb^T)) per batch, bf16 out.
// grid (64, B): 16 rows/block; wave w owns columns [w*256, w*256+256).
// ====================================================================
__global__ __launch_bounds__(128) void k_support(
    const __bf16* __restrict__ comb, __bf16* __restrict__ sup) {
  __shared__ float red[4][16];
  int w = wave_id(), l = lane_id();
  int b = blockIdx.y, i0 = blockIdx.x * 16;
  int col = l & 15, mb = (l < 16) ? 0 : 8;
  const __bf16* Cb = comb + (size_t)b * Np * Ed;
  v16bf a0 = ldA16(Cb + (size_t)i0 * Ed, Ed);
  v16bf a1 = ldA16(Cb + (size_t)i0 * Ed + 32, Ed);
  v8f sc[16];
#pragma unroll
  for (int jt = 0; jt < 16; ++jt) {
    int j0 = w * 256 + jt * 16;
    v8f acc = {};
    acc = wmma_bf(a0, ldBT16(Cb + (size_t)j0 * Ed, Ed), acc);
    acc = wmma_bf(a1, ldBT16(Cb + (size_t)j0 * Ed + 32, Ed), acc);
#pragma unroll
    for (int v = 0; v < 8; ++v) acc[v] = fmaxf(acc[v], 0.f);
    sc[jt] = acc;
  }
  float mx[8], sm[8];
#pragma unroll
  for (int v = 0; v < 8; ++v) {
    float m = -1e30f;
#pragma unroll
    for (int jt = 0; jt < 16; ++jt) m = fmaxf(m, sc[jt][v]);
#pragma unroll
    for (int m2 = 1; m2 < 16; m2 <<= 1) m = fmaxf(m, __shfl_xor(m, m2, 32));
    mx[v] = m;
  }
  if ((l & 15) == 0) {
#pragma unroll
    for (int v = 0; v < 8; ++v) red[w][(l >> 4) * 8 + v] = mx[v];
  }
  __syncthreads();
#pragma unroll
  for (int v = 0; v < 8; ++v) {
    float m = mx[v];
#pragma unroll
    for (int w2 = 0; w2 < 4; ++w2) m = fmaxf(m, red[w2][mb + v]);
    mx[v] = m;
  }
  __syncthreads();
#pragma unroll
  for (int v = 0; v < 8; ++v) {
    float s = 0.f;
#pragma unroll
    for (int jt = 0; jt < 16; ++jt) { float e = __expf(sc[jt][v] - mx[v]); sc[jt][v] = e; s += e; }
#pragma unroll
    for (int m2 = 1; m2 < 16; m2 <<= 1) s += __shfl_xor(s, m2, 32);
    sm[v] = s;
  }
  if ((l & 15) == 0) {
#pragma unroll
    for (int v = 0; v < 8; ++v) red[w][(l >> 4) * 8 + v] = sm[v];
  }
  __syncthreads();
#pragma unroll
  for (int v = 0; v < 8; ++v) {
    float s = 0.f;
#pragma unroll
    for (int w2 = 0; w2 < 4; ++w2) s += red[w2][mb + v];
    sm[v] = frcp(s);
  }
  __bf16* S = sup + ((size_t)b * Np + i0) * Np;
#pragma unroll
  for (int jt = 0; jt < 16; ++jt)
#pragma unroll
    for (int v = 0; v < 8; ++v)
      S[(size_t)(mb + v) * Np + w * 256 + jt * 16 + col] = (__bf16)(sc[jt][v] * sm[v]);
}

// ====================================================================
// Kernel 5: out = alpha * (S @ Bsrc) [+ beta * Csrc]  per batch, bf16 out.
// grid (16, B) x 128 threads. B panel staged to LDS, ping-pong buffers.
// ====================================================================
__global__ __launch_bounds__(128) void k_spmm(
    const __bf16* __restrict__ sup, const __bf16* __restrict__ Bsrc,
    const float* __restrict__ Csrc, float alpha, float beta,
    __bf16* __restrict__ out) {
  __shared__ __bf16 bst[2][32][72];
  int w = wave_id(), l = lane_id(), t = threadIdx.x;
  int b = blockIdx.y, rt0 = blockIdx.x * 64 + w * 16;
  int col = l & 15, mb = (l < 16) ? 0 : 8;
  const __bf16* Sb = sup + ((size_t)b * Np + rt0) * Np;
  const __bf16* Bb = Bsrc + (size_t)b * Np * Ed;
  v8f acc[4] = {};
  for (int ks = 0; ks < 32; ++ks) {
    int p = ks & 1;
    {
      const __bf16* s = Bb + (size_t)(ks * 32) * Ed + t * 16;
      v8bf d0 = *(const v8bf*)s;
      v8bf d1 = *(const v8bf*)(s + 8);
      __bf16* dp = &bst[p][t >> 2][(t & 3) * 16];
      *(v8bf*)dp = d0;
      *(v8bf*)(dp + 8) = d1;
    }
    if (ks + 1 < 32) __builtin_prefetch(Bb + (size_t)((ks + 1) * 32) * Ed + t * 16, 0, 0);
    __syncthreads();
    v16bf a = ldA16(Sb + ks * 32, Np);
#pragma unroll
    for (int nt = 0; nt < 4; ++nt)
      acc[nt] = wmma_bf(a, ldB_lds(&bst[p][0][nt * 16], 72), acc[nt]);
  }
  __bf16* O = out + ((size_t)b * Np + rt0) * Ed;
  const float* C = Csrc + ((size_t)b * Np + rt0) * Ed;
  bool use_c = (beta != 0.f);
#pragma unroll
  for (int nt = 0; nt < 4; ++nt)
#pragma unroll
    for (int v = 0; v < 8; ++v) {
      int rr = mb + v, c = nt * 16 + col;
      float val = alpha * acc[nt][v];
      if (use_c) val += beta * C[rr * Ed + c];
      O[rr * Ed + c] = (__bf16)val;
    }
}

// ====================================================================
// Kernel 6: out[b,n,o] = sum_{k,i} x_g[b,n,k,i]*(emb[n]@wpool)[k,i,o] + bias_n
// 64 blocks (16 nodes) x 4 waves (one 16-wide o-tile per wave).
// Per-node weight slices generated on the fly into LDS (BT layout).
// ====================================================================
__global__ __launch_bounds__(128) void k_out(
    const __bf16* __restrict__ emb_bf, const __bf16* __restrict__ wpT, // [192*64][64]
    const float* __restrict__ bias_n,
    const __bf16* __restrict__ xg0, const __bf16* __restrict__ xg1,
    const __bf16* __restrict__ xg2,
    float* __restrict__ out) {
  __shared__ __bf16 wsl[4][16][16][32]; // 64 KB: [otile][node][o][kk] (BT layout)
  int w = wave_id(), l = lane_id();
  int n0 = blockIdx.x * 16;
  int col = l & 15, mb = (l < 16) ? 0 : 8;
  v8f oacc[16] = {};
  v16bf ea0 = ldA16(emb_bf + (size_t)n0 * Ed, Ed);
  v16bf ea1 = ldA16(emb_bf + (size_t)n0 * Ed + 32, Ed);
  for (int s = 0; s < 6; ++s) {
    // weight gen: Wn[kk=tt, o] for 16 nodes; B from wpT[ki][o][d] (contiguous in d)
    for (int tt = 0; tt < 32; ++tt) {
      const __bf16* wpt = wpT + ((size_t)(s * 32 + tt) * Ed + w * 16) * Ed;
      v8f wa = {};
      wa = wmma_bf(ea0, ldBT16(wpt, Ed), wa);
      wa = wmma_bf(ea1, ldBT16(wpt + 32, Ed), wa);
#pragma unroll
      for (int v = 0; v < 8; ++v) wsl[w][mb + v][col][tt] = (__bf16)wa[v];
    }
    wait_ds();
    const __bf16* xb = (s < 2) ? xg0 : ((s < 4) ? xg1 : xg2);
    int i0 = (s & 1) * 32;
#pragma unroll
    for (int m = 0; m < 16; ++m) {
      v16bf xa = ldA16(xb + (size_t)(n0 + m) * Ed + i0, Np * Ed);
      oacc[m] = wmma_bf(xa, ldBT16(&wsl[w][m][0][0], 32), oacc[m]);
    }
    wait_ds();
  }
#pragma unroll
  for (int m = 0; m < 16; ++m)
#pragma unroll
    for (int v = 0; v < 8; ++v)
      out[(size_t)(mb + v) * (Np * Ed) + (size_t)(n0 + m) * Ed + w * 16 + col] =
          oacc[m][v] + bias_n[(size_t)(n0 + m) * Ed + w * 16 + col];
}

// ====================================================================
extern "C" void kernel_launch(void* const* d_in, const int* in_sizes, int n_in,
                              void* d_out, int out_size, void* d_ws, size_t ws_size,
                              hipStream_t stream) {
  (void)in_sizes; (void)n_in; (void)out_size; (void)ws_size;
  const float* x        = (const float*)d_in[0];
  const float* fc_w1    = (const float*)d_in[1];
  const float* fc_b1    = (const float*)d_in[2];
  const float* fc_w2    = (const float*)d_in[3];
  const float* fc_b2    = (const float*)d_in[4];
  const float* fc_w3    = (const float*)d_in[5];
  const float* fc_b3    = (const float*)d_in[6];
  const float* node_emb = (const float*)d_in[7];
  const float* Mem      = (const float*)d_in[8];
  const float* Wq       = (const float*)d_in[9];
  const float* wpool    = (const float*)d_in[10];
  const float* bias_pool= (const float*)d_in[11];
  const float* ln_g     = (const float*)d_in[12];
  const float* ln_b     = (const float*)d_in[13];
  const float* w_w      = (const float*)d_in[14];
  const float* w_b      = (const float*)d_in[15];
  const float* trans    = (const float*)d_in[16];
  const float* wr_w     = (const float*)d_in[17];
  const float* wr_b     = (const float*)d_in[18];
  const float* ur_w     = (const float*)d_in[19];
  const float* ur_b     = (const float*)d_in[20];
  const float* wh_w     = (const float*)d_in[21];
  const float* wh_b     = (const float*)d_in[22];
  const float* wu_w     = (const float*)d_in[23];
  const float* wu_b     = (const float*)d_in[24];

  // ---- workspace layout: f32 region, then bf16 region ----
  float* wsf = (float*)d_ws;
  size_t fo = 0;
  float* ne     = wsf + fo; fo += (size_t)Np * Ed;
  float* ne_res = wsf + fo; fo += (size_t)Np * Ed;
  float* u_r    = wsf + fo; fo += (size_t)Np * Ed;
  float* w_u    = wsf + fo; fo += (size_t)Np * Ed;
  float* bias_n = wsf + fo; fo += (size_t)Np * Ed;
  __bf16* wsb = (__bf16*)(wsf + fo);
  size_t bo = 0;
  __bf16* xr_bf  = wsb + bo; bo += (size_t)ROWS * Ed;   // 2 MB
  __bf16* emb_bf = wsb + bo; bo += (size_t)Np * Ed;
  __bf16* dyn_bf = wsb + bo; bo += (size_t)ROWS * Ed;
  __bf16* y1_bf  = wsb + bo; bo += (size_t)ROWS * Ed;
  __bf16* y2_bf  = wsb + bo; bo += (size_t)ROWS * Ed;
  __bf16* comb   = wsb + bo; bo += (size_t)ROWS * Ed;
  __bf16* mem_bf = wsb + bo; bo += 128 * Ed;
  __bf16* memT   = wsb + bo; bo += Ed * 128;
  __bf16* w1T    = wsb + bo; bo += Ed * Ch;
  __bf16* w2T    = wsb + bo; bo += Ch * Ch;
  __bf16* w3T    = wsb + bo; bo += Ch * Ed;
  __bf16* WqT    = wsb + bo; bo += Ed * Ed;
  __bf16* w_wT   = wsb + bo; bo += Ed * Ed;
  __bf16* ur_wT  = wsb + bo; bo += Ed * Ed;
  __bf16* wu_wT  = wsb + bo; bo += Ed * Ed;
  __bf16* transT = wsb + bo; bo += Ed * Ed;
  __bf16* wr_wT  = wsb + bo; bo += Ed * Ed;
  __bf16* wh_wT  = wsb + bo; bo += Ed * Ed;
  __bf16* bpT    = wsb + bo; bo += Ed * Ed;
  __bf16* wpT    = wsb + bo; bo += (size_t)192 * Ed * Ed; // 1.5 MB
  __bf16* sup    = wsb + bo;                              // 32 MB

  // ---- one-time conversions (bf16, BT layouts for all GEMM B operands) ----
  k_cvt  <<<(ROWS * Ed + 255) / 256, 256, 0, stream>>>(x, xr_bf, ROWS * Ed);
  k_cvt  <<<(Np * Ed + 255) / 256, 256, 0, stream>>>(node_emb, emb_bf, Np * Ed);
  k_cvt  <<<(128 * Ed + 255) / 256, 256, 0, stream>>>(Mem, mem_bf, 128 * Ed);
  k_cvt_t<<<(128 * Ed + 255) / 256, 256, 0, stream>>>(Mem, memT, 128, Ed);
  k_cvt_t<<<(Ed * Ch + 255) / 256, 256, 0, stream>>>(fc_w1, w1T, Ed, Ch);
  k_cvt_t<<<(Ch * Ch + 255) / 256, 256, 0, stream>>>(fc_w2, w2T, Ch, Ch);
  k_cvt_t<<<(Ch * Ed + 255) / 256, 256, 0, stream>>>(fc_w3, w3T, Ch, Ed);
  k_cvt_t<<<(Ed * Ed + 255) / 256, 256, 0, stream>>>(Wq, WqT, Ed, Ed);
  k_cvt_t<<<(Ed * Ed + 255) / 256, 256, 0, stream>>>(w_w, w_wT, Ed, Ed);
  k_cvt_t<<<(Ed * Ed + 255) / 256, 256, 0, stream>>>(ur_w, ur_wT, Ed, Ed);
  k_cvt_t<<<(Ed * Ed + 255) / 256, 256, 0, stream>>>(wu_w, wu_wT, Ed, Ed);
  k_cvt_t<<<(Ed * Ed + 255) / 256, 256, 0, stream>>>(trans, transT, Ed, Ed);
  k_cvt_t<<<(Ed * Ed + 255) / 256, 256, 0, stream>>>(wr_w, wr_wT, Ed, Ed);
  k_cvt_t<<<(Ed * Ed + 255) / 256, 256, 0, stream>>>(wh_w, wh_wT, Ed, Ed);
  k_cvt_t<<<(Ed * Ed + 255) / 256, 256, 0, stream>>>(bias_pool, bpT, Ed, Ed);
  k_cvt_t<<<(192 * Ed * Ed + 255) / 256, 256, 0, stream>>>(wpool, wpT, Ed, 192 * Ed);

  // ---- main pipeline ----
  k_embed<<<ROWS / 64, 128, 0, stream>>>(xr_bf, fc_b1, fc_b2, fc_b3,
                                         w1T, w2T, w3T, WqT, mem_bf, memT, dyn_bf);
  k_node <<<Np / 64, 128, 0, stream>>>(node_emb, emb_bf, ln_g, ln_b, w_wT, w_b,
                                       ur_wT, ur_b, wu_wT, wu_b, bpT,
                                       ne, ne_res, u_r, w_u, bias_n);
  k_comb <<<ROWS / 64, 128, 0, stream>>>(dyn_bf, ne, ne_res, u_r, w_u,
                                         transT, wr_wT, wr_b, wh_wT, wh_b, comb);
  k_support<<<dim3(Np / 16, Bn), 128, 0, stream>>>(comb, sup);
  // y1 = S @ xr ; y2 = 2*S@y1 - xr   (== T2 @ xr without forming S@S)
  k_spmm<<<dim3(Np / 64, Bn), 128, 0, stream>>>(sup, xr_bf, x, 1.f, 0.f, y1_bf);
  k_spmm<<<dim3(Np / 64, Bn), 128, 0, stream>>>(sup, y1_bf, x, 2.f, -1.f, y2_bf);
  k_out <<<Np / 16, 128, 0, stream>>>(emb_bf, wpT, bias_n, xr_bf, y1_bf, y2_bf,
                                      (float*)d_out);
}
